// SPALPA_72782515798233
// MI455X (gfx1250) — matrix operand order repaired and checked by reading, not compiled, not run on hardware
//
#include <hip/hip_runtime.h>
#include <math.h>

// ---------------------------------------------------------------------------
// Dims (fixed by reference): B=8, N=4096, C=64, O=128, M=16, Ns=1024, K=32
// ---------------------------------------------------------------------------
#define BB 8
#define NN 4096
#define CC 64
#define OO 128
#define MM 16
#define NS 1024
#define KS 32
#define CIN 131
#define R2 0.0225f
#define GAM 16.0f

typedef _Float16 v16h __attribute__((ext_vector_type(16)));
typedef _Float16 v8h  __attribute__((ext_vector_type(8)));
typedef float    v8f  __attribute__((ext_vector_type(8)));

// ---- WMMA helpers ----------------------------------------------------------
__device__ inline v8f wmma_f16(v16h a, v16h b, v8f c) {
  return __builtin_amdgcn_wmma_f32_16x16x32_f16(false, a, false, b, (short)0, c,
                                                false, false);
}

// A fragment (16x32 f16, rows M = lane&15; lane<16: K 0-7 & 16-23, lane>=16:
// K 8-15 & 24-31), row-major weights, row stride 64 halves.
__device__ inline v16h load_a_frag(const _Float16* W, int kbase, int lane) {
  const _Float16* r = W + (lane & 15) * 64;
  int k0 = kbase + ((lane >> 4) << 3);
  const v8h* p0 = reinterpret_cast<const v8h*>(r + k0);
  const v8h* p1 = reinterpret_cast<const v8h*>(r + k0 + 16);
  v8h lo = *p0, hi = *p1;
  v16h f;
#pragma unroll
  for (int i = 0; i < 8; ++i) { f[i] = lo[i]; f[8 + i] = hi[i]; }
  return f;
}

// B fragment (32x16 f16, K x N): cols column-major, 64 halves per column.
// lane&15 = N; lanes 0-15 hold K 0-7 & 16-23, lanes 16-31 hold K 8-15 & 24-31.
__device__ inline v16h load_b_frag(const _Float16* cols, int kbase, int lane) {
  const _Float16* c = cols + (lane & 15) * 64 + kbase + ((lane >> 4) << 3);
  const v8h* p = reinterpret_cast<const v8h*>(c);
  v8h lo = p[0], hi = p[2];
  v16h f;
#pragma unroll
  for (int i = 0; i < 8; ++i) { f[i] = lo[i]; f[8 + i] = hi[i]; }
  return f;
}

// group-of-16 reductions (wave32): xor masks 1,2,4,8 stay inside each half.
__device__ inline float red16_max(float v) {
#pragma unroll
  for (int m = 1; m < 16; m <<= 1) v = fmaxf(v, __shfl_xor(v, m, 32));
  return v;
}
__device__ inline float red16_sum(float v) {
#pragma unroll
  for (int m = 1; m < 16; m <<= 1) v += __shfl_xor(v, m, 32);
  return v;
}

// ---------------------------------------------------------------------------
// K0: build combined f16 weight matrices.
// ---------------------------------------------------------------------------
__global__ void k0_prep(const float* wc, const float* wal, const float* wg,
                        const float* wag, const float* wskip, const float* bskip,
                        _Float16* WcatL, _Float16* WcatG, _Float16* Wcorr,
                        float* bcorr) {
  int t = threadIdx.x;
  for (int e = t; e < 256 * 64; e += 256) {
    int r = e >> 6, c = e & 63;
    float vl = (r < 128) ? (wc[r * CIN + 3 + c] + wc[r * CIN + 67 + c])
                         : (wal[(r - 128) * CIN + 3 + c] + wal[(r - 128) * CIN + 67 + c]);
    WcatL[e] = (_Float16)vl;
    float vg = (r < 128) ? (wg[r * CIN + 3 + c] + wg[r * CIN + 67 + c])
                         : (wag[(r - 128) * CIN + 3 + c] + wag[(r - 128) * CIN + 67 + c]);
    WcatG[e] = (_Float16)vg;
  }
  for (int e = t; e < 640 * 64; e += 256) {
    int r = e >> 6, c = e & 63;
    float v;
    if (r < 128)      v =  wskip[r * 64 + c];
    else if (r < 256) v = -wc [(r - 128) * CIN + 67 + c];
    else if (r < 384) v = -wal[(r - 256) * CIN + 67 + c];
    else if (r < 512) v = -wg [(r - 384) * CIN + 67 + c];
    else              v = -wag[(r - 512) * CIN + 67 + c];
    Wcorr[e] = (_Float16)v;
  }
  for (int r = t; r < 640; r += 256) bcorr[r] = (r < 128) ? bskip[r] : 0.f;
}

// ---------------------------------------------------------------------------
// K1: farthest point sampling (sequential, one block per batch)
// ---------------------------------------------------------------------------
__global__ void k1_fps(const float* p, int* idx) {
  __shared__ float sd[NN];
  __shared__ float rv[256];
  __shared__ int   ri[256];
  __shared__ int   sLast;
  int b = blockIdx.x, t = threadIdx.x;
  for (int i = t; i < NN; i += 256) sd[i] = 1e10f;
  if (t == 0) { sLast = 0; idx[b * NS] = 0; }
  __syncthreads();
  const float* pb = p + (size_t)b * NN * 3;
  for (int step = 1; step < NS; ++step) {
    int last = sLast;
    float lx = pb[last * 3 + 0], ly = pb[last * 3 + 1], lz = pb[last * 3 + 2];
    float bv = -1.f; int bi = 0;
    for (int i = t; i < NN; i += 256) {
      float dx = pb[i * 3 + 0] - lx, dy = pb[i * 3 + 1] - ly, dz = pb[i * 3 + 2] - lz;
      float d = fminf(sd[i], dx * dx + dy * dy + dz * dz);
      sd[i] = d;
      if (d > bv) { bv = d; bi = i; }
    }
    rv[t] = bv; ri[t] = bi;
    __syncthreads();
    for (int s = 128; s > 0; s >>= 1) {
      if (t < s) {
        if (rv[t + s] > rv[t] || (rv[t + s] == rv[t] && ri[t + s] < ri[t])) {
          rv[t] = rv[t + s]; ri[t] = ri[t + s];
        }
      }
      __syncthreads();
    }
    if (t == 0) { sLast = ri[0]; idx[b * NS + step] = ri[0]; }
    __syncthreads();
  }
}

// K2a: f (B,C,N) f32 -> f_t (B,N,C) f16
__global__ void k2a_transpose(const float* f, _Float16* f_t) {
  int e = blockIdx.x * 256 + threadIdx.x;
  if (e >= BB * CC * NN) return;
  int n = e & (NN - 1), c = (e >> 12) & 63, b = e >> 18;
  f_t[((size_t)(b * NN + n)) * 64 + c] = (_Float16)f[e];
}

// K2b: new_p gather -> output region 0
__global__ void k2b_newp(const float* p, const int* idx, float* newp) {
  int t = blockIdx.x * 256 + threadIdx.x;
  if (t >= BB * NS) return;
  int b = t >> 10; int gi = idx[t];
  const float* s = p + ((size_t)b * NN + gi) * 3;
  newp[t * 3 + 0] = s[0]; newp[t * 3 + 1] = s[1]; newp[t * 3 + 2] = s[2];
}

// K2c: fi (B,64,1024) f32 gather
__global__ void k2c_fi(const float* f, const int* idx, float* fi) {
  int e = blockIdx.x * 256 + threadIdx.x;
  if (e >= BB * CC * NS) return;
  int n = e & 1023, c = (e >> 10) & 63, b = e >> 16;
  fi[e] = f[((size_t)(b * CC + c)) * NN + idx[b * NS + n]];
}

// ---------------------------------------------------------------------------
// K3: ball query — one wave32 per query, ordered index compaction via ballot
// ---------------------------------------------------------------------------
__global__ void k3_ball(const float* p, const float* newp, int* gidx) {
  int wid  = (blockIdx.x * 256 + threadIdx.x) >> 5;
  int lane = threadIdx.x & 31;
  int b = wid >> 10;
  const float* q = newp + (size_t)wid * 3;
  float qx = q[0], qy = q[1], qz = q[2];
  const float* pb = p + (size_t)b * NN * 3;
  int* g = gidx + (size_t)wid * KS;
  int cnt = 0, first = -1;
  for (int base = 0; base < NN; base += 32) {
    int i = base + lane;
    float dx = pb[i * 3 + 0] - qx, dy = pb[i * 3 + 1] - qy, dz = pb[i * 3 + 2] - qz;
    bool ok = (dx * dx + dy * dy + dz * dz) <= R2;
    unsigned mask = __builtin_amdgcn_ballot_w32(ok);
    if (first < 0 && mask) first = base + __builtin_ctz(mask);
    if (ok) {
      int pos = cnt + __builtin_popcount(mask & ((1u << lane) - 1u));
      if (pos < KS) g[pos] = i;
    }
    cnt += __builtin_popcount(mask);
    if (cnt >= KS) break;
  }
  if (cnt < KS) {
    if (first < 0) first = 0;
    if (lane >= cnt) g[lane] = first;
  }
}

// ---------------------------------------------------------------------------
// K4: corr GEMM (WMMA): corr[b,n,:] = Wcorr(640x64) @ fi_f16(:,n) + bcorr
// corr layout (B, NS, 640) -> lane-contiguous b128 stores + coalesced readers
// ---------------------------------------------------------------------------
__global__ void k4_corr(const _Float16* f_t, const int* idx, const _Float16* Wcorr,
                        const float* bcorr, float* corr) {
  __shared__ __align__(16) _Float16 sFi[64 * 64];
  int b = blockIdx.x >> 4;
  int n0 = (blockIdx.x & 15) * 64;
  int t = threadIdx.x;
  {
    int col = t & 63, qq = t >> 6;
    int gi = idx[b * NS + n0 + col];
    const _Float16* src = f_t + ((size_t)(b * NN + gi)) * 64 + qq * 16;
    _Float16* dst = sFi + col * 64 + qq * 16;
#pragma unroll
    for (int i = 0; i < 16; ++i) dst[i] = src[i];
  }
  __syncthreads();
  int wave = t >> 5, lane = t & 31, hi = lane >> 4;
  const v8f z8 = {0.f, 0.f, 0.f, 0.f, 0.f, 0.f, 0.f, 0.f};
#pragma unroll 1
  for (int j = 0; j < 5; ++j) {
    int rt = wave * 5 + j;
    const _Float16* Wr = Wcorr + rt * 16 * 64;
    v16h a0 = load_a_frag(Wr, 0, lane);
    v16h a1 = load_a_frag(Wr, 32, lane);
    float bcv[8];
#pragma unroll
    for (int v = 0; v < 8; ++v) bcv[v] = bcorr[rt * 16 + 8 * hi + v];
#pragma unroll 1
    for (int cp = 0; cp < 2; ++cp) {
      const _Float16* colA = sFi + (cp * 2 + 0) * 16 * 64;
      const _Float16* colB = sFi + (cp * 2 + 1) * 16 * 64;
      v16h bA0 = load_b_frag(colA, 0, lane);
      v16h bB0 = load_b_frag(colB, 0, lane);
      v16h bA1 = load_b_frag(colA, 32, lane);
      v16h bB1 = load_b_frag(colB, 32, lane);
      // independent WMMAs first, then the dependent accumulations
      v8f accA = wmma_f16(a0, bA0, z8);
      v8f accB = wmma_f16(a0, bB0, z8);
      accA = wmma_f16(a1, bA1, accA);
      accB = wmma_f16(a1, bB1, accB);
      int nA = n0 + (cp * 2) * 16 + (lane & 15);
      int nB = nA + 16;
      float* dA = corr + ((size_t)(b * NS) + nA) * 640 + rt * 16 + 8 * hi;
      float* dB = corr + ((size_t)(b * NS) + nB) * 640 + rt * 16 + 8 * hi;
      reinterpret_cast<float4*>(dA)[0] =
          make_float4(accA[0] + bcv[0], accA[1] + bcv[1], accA[2] + bcv[2], accA[3] + bcv[3]);
      reinterpret_cast<float4*>(dA)[1] =
          make_float4(accA[4] + bcv[4], accA[5] + bcv[5], accA[6] + bcv[6], accA[7] + bcv[7]);
      reinterpret_cast<float4*>(dB)[0] =
          make_float4(accB[0] + bcv[0], accB[1] + bcv[1], accB[2] + bcv[2], accB[3] + bcv[3]);
      reinterpret_cast<float4*>(dB)[1] =
          make_float4(accB[4] + bcv[4], accB[5] + bcv[5], accB[6] + bcv[6], accB[7] + bcv[7]);
    }
  }
}

// ---------------------------------------------------------------------------
// Kg1: imap row + softmax over n + global_p + attn3; one block per (b,m)
// ---------------------------------------------------------------------------
__global__ void kg1_global(const float* z, const float* fi, const float* newp,
                           float* attn3, float* gp) {
  __shared__ float sZ[64];
  __shared__ float sRow[NS];
  __shared__ float sRed[256];
  __shared__ float sGP[3];
  int b = blockIdx.x >> 4, m = blockIdx.x & 15, t = threadIdx.x;
  if (t < 64) sZ[t] = z[m * 64 + t];
  __syncthreads();
  const float* fib = fi + (size_t)b * CC * NS;
  float vloc[4];
#pragma unroll
  for (int j = 0; j < 4; ++j) {
    int n = t + 256 * j;
    float acc = 0.f;
    for (int c = 0; c < 64; ++c) acc += sZ[c] * fib[c * NS + n];
    sRow[n] = acc; vloc[j] = acc;
  }
  float mx = fmaxf(fmaxf(vloc[0], vloc[1]), fmaxf(vloc[2], vloc[3]));
  sRed[t] = mx; __syncthreads();
  for (int s = 128; s > 0; s >>= 1) { if (t < s) sRed[t] = fmaxf(sRed[t], sRed[t + s]); __syncthreads(); }
  mx = sRed[0]; __syncthreads();
  float lsum = 0.f;
#pragma unroll
  for (int j = 0; j < 4; ++j) { int n = t + 256 * j; float e = __expf(sRow[n] - mx); sRow[n] = e; lsum += e; }
  sRed[t] = lsum; __syncthreads();
  for (int s = 128; s > 0; s >>= 1) { if (t < s) sRed[t] += sRed[t + s]; __syncthreads(); }
  float inv = 1.f / sRed[0]; __syncthreads();
  float gpl[3] = {0.f, 0.f, 0.f};
#pragma unroll
  for (int j = 0; j < 4; ++j) {
    int n = t + 256 * j;
    float sft = sRow[n] * inv; sRow[n] = sft;
    const float* np = newp + ((size_t)(b * NS) + n) * 3;
    gpl[0] += sft * np[0]; gpl[1] += sft * np[1]; gpl[2] += sft * np[2];
  }
  for (int d = 0; d < 3; ++d) {
    sRed[t] = gpl[d]; __syncthreads();
    for (int s = 128; s > 0; s >>= 1) { if (t < s) sRed[t] += sRed[t + s]; __syncthreads(); }
    if (t == 0) { sGP[d] = sRed[0]; gp[(b * MM + m) * 3 + d] = sRed[0]; }
    __syncthreads();
  }
  float gx = sGP[0], gy = sGP[1], gz = sGP[2];
#pragma unroll
  for (int j = 0; j < 4; ++j) {
    int n = t + 256 * j;
    const float* np = newp + ((size_t)(b * NS) + n) * 3;
    float dx = gx - np[0], dy = gy - np[1], dz = gz - np[2];
    float d2 = dx * dx + dy * dy + dz * dz;
    attn3[((size_t)(b * MM + m)) * NS + n] = sRow[n] * __expf(-GAM * d2);
  }
}

// Kg3: global_f[b,c,m] = sum_n fi[c,n]*attn3[m,n]; one block per (b,m), 64 thr
__global__ void kg3_gf(const float* fi, const float* attn3, float* gf) {
  __shared__ float sA[NS];
  int b = blockIdx.x >> 4, m = blockIdx.x & 15, t = threadIdx.x;
  for (int j = t; j < NS; j += 64) sA[j] = attn3[((size_t)(b * MM + m)) * NS + j];
  __syncthreads();
  const float* fr = fi + ((size_t)(b * CC + t)) * NS;
  float acc = 0.f;
  for (int n = 0; n < NS; ++n) acc += fr[n] * sA[n];
  gf[(b * CC + t) * MM + m] = acc;
}

// Kg4: wsG[b] = WcatG(256x64) @ global_f(64x16)  (WMMA), one block per b
__global__ void kg4_G(const _Float16* WcatG, const float* gf, float* wsG) {
  __shared__ __align__(16) _Float16 sGF[16 * 64];
  int b = blockIdx.x, t = threadIdx.x;
  for (int e = t; e < 1024; e += 256) {
    int m = e >> 6, c = e & 63;
    sGF[m * 64 + c] = (_Float16)gf[(b * CC + c) * MM + m];
  }
  __syncthreads();
  int wave = t >> 5, lane = t & 31, hi = lane >> 4;
  const v8f z8 = {0.f, 0.f, 0.f, 0.f, 0.f, 0.f, 0.f, 0.f};
  v16h b0 = load_b_frag(sGF, 0, lane);
  v16h b1 = load_b_frag(sGF, 32, lane);
  for (int j = 0; j < 2; ++j) {
    int rt = wave * 2 + j;
    const _Float16* Wr = WcatG + rt * 16 * 64;
    v16h a0 = load_a_frag(Wr, 0, lane);
    v16h a1 = load_a_frag(Wr, 32, lane);
    v8f acc = wmma_f16(a0, b0, z8);
    acc = wmma_f16(a1, b1, acc);
    int m = lane & 15;
#pragma unroll
    for (int v = 0; v < 8; ++v) {
      int row = rt * 16 + v + 8 * hi;
      wsG[((size_t)(b * 256) + row) * MM + m] = acc[v];
    }
  }
}

// ---------------------------------------------------------------------------
// K5: fused local branch (async global->LDS staging, WMMA, in-register
// softmax over k=32). block = (b, 8 queries); 8 waves = 8 o-tiles.
// Epilogue per-row constants live in LDS (sEp) to keep VGPR count low.
// ---------------------------------------------------------------------------
__global__ void k5_local(const float* p, const float* newp, const int* gidx,
                         const _Float16* f_t, const _Float16* WcatL,
                         const float* corr, const float* wc, const float* sc_g,
                         const float* bc_g, const float* wal, const float* sa_g,
                         const float* ba_g, float* localbuf) {
  __shared__ __align__(16) _Float16 sF[256 * 64];
  __shared__ float sDP[3][256];
  __shared__ __align__(16) float sEp[128][12];
  int b = blockIdx.x >> 7;
  int n0 = (blockIdx.x & 127) * 8;
  int t = threadIdx.x;
  {
    int col = t;
    int n = col >> 5, k = col & 31;
    int gi = gidx[(((size_t)b * NS) + n0 + n) * KS + k];
    // async DMA: 128 bytes of gathered f16 feature row directly into LDS.
    // INST_OFFSET is added to both the LDS and global addresses (ISA 15.18.3),
    // so one base pair serves all 8 b128 transfers.
    unsigned lds = (unsigned)(uintptr_t)(sF + col * 64);
    unsigned long long ga =
        (unsigned long long)(uintptr_t)(f_t + ((size_t)(b * NN) + gi) * 64);
    asm volatile(
        "global_load_async_to_lds_b128 %0, %1, off\n\t"
        "global_load_async_to_lds_b128 %0, %1, off offset:16\n\t"
        "global_load_async_to_lds_b128 %0, %1, off offset:32\n\t"
        "global_load_async_to_lds_b128 %0, %1, off offset:48\n\t"
        "global_load_async_to_lds_b128 %0, %1, off offset:64\n\t"
        "global_load_async_to_lds_b128 %0, %1, off offset:80\n\t"
        "global_load_async_to_lds_b128 %0, %1, off offset:96\n\t"
        "global_load_async_to_lds_b128 %0, %1, off offset:112"
        :: "v"(lds), "v"(ga) : "memory");
    const float* ps = p + ((size_t)(b * NN) + gi) * 3;
    const float* qs = newp + ((size_t)(b * NS) + n0 + n) * 3;
    sDP[0][col] = ps[0] - qs[0];
    sDP[1][col] = ps[1] - qs[1];
    sDP[2][col] = ps[2] - qs[2];
  }
  if (t < 128) {  // per-row epilogue constants -> LDS
    sEp[t][0] = sc_g[t]; sEp[t][1] = bc_g[t];
    sEp[t][2] = sa_g[t]; sEp[t][3] = ba_g[t];
    sEp[t][4] = wc[t * CIN + 0];  sEp[t][5] = wc[t * CIN + 1];
    sEp[t][6] = wc[t * CIN + 2];  sEp[t][7] = 0.f;
    sEp[t][8] = wal[t * CIN + 0]; sEp[t][9] = wal[t * CIN + 1];
    sEp[t][10] = wal[t * CIN + 2]; sEp[t][11] = 0.f;
  }
  asm volatile("s_wait_asynccnt 0x0" ::: "memory");
  __syncthreads();
  int wave = t >> 5, lane = t & 31, hi = lane >> 4, ln = lane & 15;
  const _Float16* Wx = WcatL + wave * 16 * 64;
  const _Float16* Wa = WcatL + (128 + wave * 16) * 64;
  v16h ax0 = load_a_frag(Wx, 0, lane);
  v16h ax1 = load_a_frag(Wx, 32, lane);
  v16h aa0 = load_a_frag(Wa, 0, lane);
  v16h aa1 = load_a_frag(Wa, 32, lane);
  const v8f z8 = {0.f, 0.f, 0.f, 0.f, 0.f, 0.f, 0.f, 0.f};
#pragma unroll 1
  for (int n = 0; n < 8; ++n) {
    const _Float16* c0 = sF + (n * 32) * 64;
    const _Float16* c1 = sF + (n * 32 + 16) * 64;
    v16h b00 = load_b_frag(c0, 0, lane);
    v16h b10 = load_b_frag(c1, 0, lane);
    v16h b01 = load_b_frag(c0, 32, lane);
    v16h b11 = load_b_frag(c1, 32, lane);
    // 4 independent WMMAs, then 4 dependent accumulations (hides RAW hazard)
    v8f accX0 = wmma_f16(ax0, b00, z8);
    v8f accX1 = wmma_f16(ax0, b10, z8);
    v8f accA0 = wmma_f16(aa0, b00, z8);
    v8f accA1 = wmma_f16(aa0, b10, z8);
    accX0 = wmma_f16(ax1, b01, accX0);
    accX1 = wmma_f16(ax1, b11, accX1);
    accA0 = wmma_f16(aa1, b01, accA0);
    accA1 = wmma_f16(aa1, b11, accA1);
    int nn = n0 + n;
    // per-n correction rows (contiguous float4 pairs)
    const float* cb = corr + ((size_t)(b * NS) + nn) * 640 + wave * 16 + 8 * hi;
    float4 cxa = reinterpret_cast<const float4*>(cb + 128)[0];
    float4 cxb = reinterpret_cast<const float4*>(cb + 128)[1];
    float4 caa = reinterpret_cast<const float4*>(cb + 256)[0];
    float4 cab = reinterpret_cast<const float4*>(cb + 256)[1];
    float cxv[8] = {cxa.x, cxa.y, cxa.z, cxa.w, cxb.x, cxb.y, cxb.z, cxb.w};
    float cav[8] = {caa.x, caa.y, caa.z, caa.w, cab.x, cab.y, cab.z, cab.w};
    int col0 = n * 32 + ln, col1 = col0 + 16;
    float dp00 = sDP[0][col0], dp01 = sDP[1][col0], dp02 = sDP[2][col0];
    float dp10 = sDP[0][col1], dp11 = sDP[1][col1], dp12 = sDP[2][col1];
    float res[8];
#pragma unroll
    for (int v = 0; v < 8; ++v) {
      const float* ep = &sEp[wave * 16 + 8 * hi + v][0];
      float4 e0 = reinterpret_cast<const float4*>(ep)[0];  // sc,bc,sa,ba
      float4 e1 = reinterpret_cast<const float4*>(ep)[1];  // wpx0..2
      float4 e2 = reinterpret_cast<const float4*>(ep)[2];  // wpa0..2
      float px0 = e1.x * dp00 + e1.y * dp01 + e1.z * dp02;
      float px1 = e1.x * dp10 + e1.y * dp11 + e1.z * dp12;
      float pa0 = e2.x * dp00 + e2.y * dp01 + e2.z * dp02;
      float pa1 = e2.x * dp10 + e2.y * dp11 + e2.z * dp12;
      float x0 = e0.x * (accX0[v] + cxv[v] + px0) + e0.y;
      float x1 = e0.x * (accX1[v] + cxv[v] + px1) + e0.y;
      float a0 = e0.z * (accA0[v] + cav[v] + pa0) + e0.w;
      float a1 = e0.z * (accA1[v] + cav[v] + pa1) + e0.w;
      float mxv = red16_max(fmaxf(a0, a1));
      float e_0 = __expf(a0 - mxv), e_1 = __expf(a1 - mxv);
      float ssum = red16_sum(e_0 + e_1);
      float num  = red16_sum(e_0 * x0 + e_1 * x1);
      res[v] = num / ssum;
    }
    if (ln == 0) {
      float* d = localbuf + ((size_t)(b * NS) + nn) * 128 + wave * 16 + 8 * hi;
      reinterpret_cast<float4*>(d)[0] = make_float4(res[0], res[1], res[2], res[3]);
      reinterpret_cast<float4*>(d)[1] = make_float4(res[4], res[5], res[6], res[7]);
    }
  }
}

// ---------------------------------------------------------------------------
// K6: global branch epilogue (softmax over m=16 in-thread) + blend + relu
// ---------------------------------------------------------------------------
__global__ void k6_final(const float* wsG, const float* corr, const float* localbuf,
                         const float* gp, const float* newp,
                         const float* wg, const float* sg_g, const float* bg_g,
                         const float* wag, const float* sag_g, const float* bag_g,
                         const float* alpha, float* fout) {
  __shared__ float sGP[48];
  __shared__ float sNP[3];
  int b = blockIdx.x >> 10, n = blockIdx.x & 1023, o = threadIdx.x;
  if (o < 48) sGP[o] = gp[b * 48 + o];
  if (o >= 48 && o < 51) sNP[o - 48] = newp[((size_t)(b * NS) + n) * 3 + (o - 48)];
  __syncthreads();
  const float* gxp = wsG + ((size_t)(b * 256) + o) * MM;
  const float* gap = wsG + ((size_t)(b * 256) + 128 + o) * MM;
  const float* cb = corr + ((size_t)(b * NS) + n) * 640;
  float cx = cb[384 + o];
  float ca = cb[512 + o];
  float ident = cb[o];
  float sg = sg_g[o], bg = bg_g[o], sa = sag_g[o], ba = bag_g[o];
  float wg0 = wg[o * CIN + 0], wg1 = wg[o * CIN + 1], wg2 = wg[o * CIN + 2];
  float wa0 = wag[o * CIN + 0], wa1 = wag[o * CIN + 1], wa2 = wag[o * CIN + 2];
  float nx = sNP[0], ny = sNP[1], nz = sNP[2];
  float xg[16], ag[16];
  float amax = -1e30f;
#pragma unroll
  for (int m = 0; m < 16; ++m) {
    float dx = sGP[m * 3 + 0] - nx, dy = sGP[m * 3 + 1] - ny, dz = sGP[m * 3 + 2] - nz;
    float xv = sg * (gxp[m] + cx + wg0 * dx + wg1 * dy + wg2 * dz) + bg;
    float av = sa * (gap[m] + ca + wa0 * dx + wa1 * dy + wa2 * dz) + ba;
    xg[m] = xv; ag[m] = av; amax = fmaxf(amax, av);
  }
  float ssum = 0.f, num = 0.f;
#pragma unroll
  for (int m = 0; m < 16; ++m) { float e = __expf(ag[m] - amax); ssum += e; num += e * xg[m]; }
  float gout = num / ssum;
  float aa = 1.f / (1.f + __expf(-alpha[0]));
  float lf = localbuf[((size_t)(b * NS) + n) * 128 + o];
  float val = lf * (1.f - aa) + gout * aa + ident;
  fout[((size_t)(b * OO) + o) * NS + n] = fmaxf(val, 0.f);
}

// ---------------------------------------------------------------------------
extern "C" void kernel_launch(void* const* d_in, const int* in_sizes, int n_in,
                              void* d_out, int out_size, void* d_ws, size_t ws_size,
                              hipStream_t stream) {
  const float* p        = (const float*)d_in[0];
  const float* f        = (const float*)d_in[1];
  const float* w_convs  = (const float*)d_in[2];
  const float* s_convs  = (const float*)d_in[3];
  const float* b_convs  = (const float*)d_in[4];
  const float* w_attn_l = (const float*)d_in[5];
  const float* s_attn_l = (const float*)d_in[6];
  const float* b_attn_l = (const float*)d_in[7];
  const float* w_gconvs = (const float*)d_in[8];
  const float* s_gconvs = (const float*)d_in[9];
  const float* b_gconvs = (const float*)d_in[10];
  const float* w_attn_g = (const float*)d_in[11];
  const float* s_attn_g = (const float*)d_in[12];
  const float* b_attn_g = (const float*)d_in[13];
  const float* w_skip   = (const float*)d_in[14];
  const float* b_skip   = (const float*)d_in[15];
  const float* z        = (const float*)d_in[16];
  const float* alpha    = (const float*)d_in[17];

  char* ws = (char*)d_ws;
  size_t off = 0;
  auto take = [&](size_t bytes) { size_t o = off; off = (off + bytes + 255) & ~(size_t)255; return o; };
  int*      idx      = (int*)     (ws + take((size_t)BB * NS * 4));
  int*      gidx     = (int*)     (ws + take((size_t)BB * NS * KS * 4));
  _Float16* f_t      = (_Float16*)(ws + take((size_t)BB * NN * CC * 2));
  float*    fi       = (float*)   (ws + take((size_t)BB * CC * NS * 4));
  _Float16* WcatL    = (_Float16*)(ws + take(256 * 64 * 2));
  _Float16* WcatG    = (_Float16*)(ws + take(256 * 64 * 2));
  _Float16* Wcorr    = (_Float16*)(ws + take(640 * 64 * 2));
  float*    bcorr    = (float*)   (ws + take(640 * 4));
  float*    corr     = (float*)   (ws + take((size_t)BB * NS * 640 * 4));
  float*    localbuf = (float*)   (ws + take((size_t)BB * NS * OO * 4));
  float*    attn3    = (float*)   (ws + take((size_t)BB * MM * NS * 4));
  float*    gp       = (float*)   (ws + take((size_t)BB * MM * 3 * 4));
  float*    gf       = (float*)   (ws + take((size_t)BB * CC * MM * 4));
  float*    wsG      = (float*)   (ws + take((size_t)BB * 256 * MM * 4));
  // total ~33.4 MB

  float* newp = (float*)d_out;              // (B,Ns,3) — first output
  float* fout = newp + (size_t)BB * NS * 3; // (B,O,Ns) — second output

  k0_prep<<<1, 256, 0, stream>>>(w_convs, w_attn_l, w_gconvs, w_attn_g, w_skip,
                                 b_skip, WcatL, WcatG, Wcorr, bcorr);
  k2a_transpose<<<(BB * CC * NN) / 256, 256, 0, stream>>>(f, f_t);
  k1_fps<<<BB, 256, 0, stream>>>(p, idx);
  k2b_newp<<<(BB * NS) / 256, 256, 0, stream>>>(p, idx, newp);
  k2c_fi<<<(BB * CC * NS) / 256, 256, 0, stream>>>(f, idx, fi);
  k3_ball<<<(BB * NS) / 8, 256, 0, stream>>>(p, newp, gidx);
  k4_corr<<<BB * 16, 256, 0, stream>>>(f_t, idx, Wcorr, bcorr, corr);
  kg1_global<<<BB * MM, 256, 0, stream>>>(z, fi, newp, attn3, gp);
  kg3_gf<<<BB * MM, 64, 0, stream>>>(fi, attn3, gf);
  kg4_G<<<BB, 256, 0, stream>>>(WcatG, gf, wsG);
  k5_local<<<BB * (NS / 8), 256, 0, stream>>>(p, newp, gidx, f_t, WcatL, corr,
                                              w_convs, s_convs, b_convs,
                                              w_attn_l, s_attn_l, b_attn_l,
                                              localbuf);
  k6_final<<<BB * NS, 128, 0, stream>>>(wsG, corr, localbuf, gp, newp, w_gconvs,
                                        s_gconvs, b_gconvs, w_attn_g, s_attn_g,
                                        b_attn_g, alpha, fout);
}
